// OT_Alignment_10634339025547
// MI455X (gfx1250) — matrix operand order repaired and checked by reading, not compiled, not run on hardware
//
#include <hip/hip_runtime.h>
#include <math.h>

typedef __attribute__((ext_vector_type(8)))  float  v8f;
typedef __attribute__((ext_vector_type(8)))  __bf16 v8bf;
typedef __attribute__((ext_vector_type(16))) __bf16 v16bf;

#define DIM      512
#define EPS      0.05f
#define EPS_INV  20.0f   // 1/0.05
#define KTILE    32
#define LSTR     40      // LDS row stride in bf16 elements (80B: conflict-free)

// ---------------- reductions (wave32) ----------------
__device__ __forceinline__ float waveReduce(float v) {
#pragma unroll
  for (int o = 16; o > 0; o >>= 1) v += __shfl_xor(v, o, 32);
  return v;
}

__device__ __forceinline__ float blockReduce256(float v) {
  __shared__ float s[8];
  v = waveReduce(v);
  const int lane = threadIdx.x & 31, w = threadIdx.x >> 5;
  if (lane == 0) s[w] = v;
  __syncthreads();
  v = (threadIdx.x < 8) ? s[threadIdx.x] : 0.0f;
  if (w == 0) v = waveReduce(v);
  return v;  // valid on thread 0
}

// ---------------- gfx1250 async global->LDS copy helpers ----------------
__device__ __forceinline__ void async_copy16(unsigned lds_off, unsigned long long base,
                                             unsigned byte_off) {
  asm volatile("global_load_async_to_lds_b128 %0, %1, %2"
               :: "v"(lds_off), "v"(byte_off), "s"(base)
               : "memory");
}
__device__ __forceinline__ void wait_async0() {
  asm volatile("s_wait_asynccnt 0x0" ::: "memory");
}
__device__ __forceinline__ unsigned lds_addr_of(const void* p) {
  return (unsigned)(unsigned long long)p;  // low 32 bits of generic addr == LDS offset
}

// ---------------- row L2 normalize + bf16 hi/lo split ----------------
// blocks [0,N) -> xa planes, [N,2N) -> xt planes
__global__ void __launch_bounds__(256)
l2norm_split(const float* __restrict__ xa, const float* __restrict__ xt,
             __bf16* __restrict__ xa_hi, __bf16* __restrict__ xa_lo,
             __bf16* __restrict__ xt_hi, __bf16* __restrict__ xt_lo, int n) {
  const int b = blockIdx.x;
  const float* src;
  __bf16 *dh, *dl;
  if (b < n) {
    src = xa + (size_t)b * DIM;
    dh = xa_hi + (size_t)b * DIM; dl = xa_lo + (size_t)b * DIM;
  } else {
    src = xt + (size_t)(b - n) * DIM;
    dh = xt_hi + (size_t)(b - n) * DIM; dl = xt_lo + (size_t)(b - n) * DIM;
  }
  const float x0 = src[threadIdx.x];
  const float x1 = src[threadIdx.x + 256];
  const float ss = blockReduce256(x0 * x0 + x1 * x1);
  __shared__ float scale_s;
  if (threadIdx.x == 0) scale_s = 1.0f / fmaxf(sqrtf(ss), 1e-12f);
  __syncthreads();
  const float sc = scale_s;
  const float y0 = x0 * sc, y1 = x1 * sc;
  const __bf16 h0 = (__bf16)y0, h1 = (__bf16)y1;
  dh[threadIdx.x]       = h0;
  dh[threadIdx.x + 256] = h1;
  dl[threadIdx.x]       = (__bf16)(y0 - (float)h0);
  dl[threadIdx.x + 256] = (__bf16)(y1 - (float)h1);
}

__global__ void init_uv(float* __restrict__ u, float* __restrict__ v, int n) {
  const int i = blockIdx.x * 256 + threadIdx.x;
  if (i < n) {
    const float x = 1.0f / (float)n;
    u[i] = x;
    v[i] = x;
  }
}

// ---------------- GEMM (split-bf16, 3 products) + fused Gibbs: K = exp(20*(s-1)) -------------
// Block tile 128x128, 8 waves in 4(M) x 2(N), wave tile 32x64 = 2x4 accumulators.
// LDS: 4 panels (Ahi, Alo, Bhi, Blo) of 128 x KTILE bf16, stride 40 (conflict-free).
__global__ void __launch_bounds__(256)
gemm_exp_wmma(const __bf16* __restrict__ Ah_g, const __bf16* __restrict__ Al_g,
              const __bf16* __restrict__ Bh_g, const __bf16* __restrict__ Bl_g,
              float* __restrict__ K, int n) {
  __shared__ __align__(16) __bf16 Ah[128 * LSTR];
  __shared__ __align__(16) __bf16 Al[128 * LSTR];
  __shared__ __align__(16) __bf16 Bh[128 * LSTR];
  __shared__ __align__(16) __bf16 Bl[128 * LSTR];

  const int tid  = threadIdx.x;
  const int lane = tid & 31;
  const int wave = tid >> 5;
  const int half = lane >> 4;
  const int l16  = lane & 15;
  const int wm   = wave & 3;    // 4 waves over M
  const int wn   = wave >> 2;   // 2 waves over N
  const int rowBase = wm * 32;
  const int colBase = wn * 64;
  const size_t gRow0 = (size_t)blockIdx.y * 128;
  const size_t gCol0 = (size_t)blockIdx.x * 128;

  const unsigned ah0 = lds_addr_of(&Ah[0]);
  const unsigned al0 = lds_addr_of(&Al[0]);
  const unsigned bh0 = lds_addr_of(&Bh[0]);
  const unsigned bl0 = lds_addr_of(&Bl[0]);

  // per-thread staging coordinates: 512 16B-chunks per panel, 2 per thread
  const int idx0 = tid, idx1 = tid + 256;
  const int r0c = idx0 >> 2, c0c = (idx0 & 3) << 3;  // row, bf16-col (0,8,16,24)
  const int r1c = idx1 >> 2, c1c = (idx1 & 3) << 3;

  v8f acc[2][4] = {};

  for (int kb = 0; kb < DIM; kb += KTILE) {
    // ---- async stage 4 panels (safe: previous iteration ended with a barrier) ----
    const unsigned gA0 = (unsigned)(((gRow0 + r0c) * DIM + kb + c0c) * 2);
    const unsigned gA1 = (unsigned)(((gRow0 + r1c) * DIM + kb + c1c) * 2);
    const unsigned gB0 = (unsigned)(((gCol0 + r0c) * DIM + kb + c0c) * 2);
    const unsigned gB1 = (unsigned)(((gCol0 + r1c) * DIM + kb + c1c) * 2);
    const unsigned l0  = (unsigned)((r0c * LSTR + c0c) * 2);
    const unsigned l1  = (unsigned)((r1c * LSTR + c1c) * 2);
    async_copy16(ah0 + l0, (unsigned long long)Ah_g, gA0);
    async_copy16(ah0 + l1, (unsigned long long)Ah_g, gA1);
    async_copy16(al0 + l0, (unsigned long long)Al_g, gA0);
    async_copy16(al0 + l1, (unsigned long long)Al_g, gA1);
    async_copy16(bh0 + l0, (unsigned long long)Bh_g, gB0);
    async_copy16(bh0 + l1, (unsigned long long)Bh_g, gB1);
    async_copy16(bl0 + l0, (unsigned long long)Bl_g, gB0);
    async_copy16(bl0 + l1, (unsigned long long)Bl_g, gB1);
    wait_async0();
    __syncthreads();

    // ---- fragments from LDS ----
    // A (16-bit 16x32 layout): lane half=0 -> K 0..7 & 16..23 ; half=1 -> K 8..15 & 24..31
    v16bf a_h[2], a_l[2], b_h[4], b_l[4];
#pragma unroll
    for (int i = 0; i < 2; ++i) {
      const int ro = (rowBase + i * 16 + l16) * LSTR;
      {
        v8bf c0 = *(const v8bf*)(&Ah[ro + 8 * half]);
        v8bf c1 = *(const v8bf*)(&Ah[ro + 16 + 8 * half]);
        a_h[i] = __builtin_shufflevector(c0, c1, 0,1,2,3,4,5,6,7,8,9,10,11,12,13,14,15);
      }
      {
        v8bf c0 = *(const v8bf*)(&Al[ro + 8 * half]);
        v8bf c1 = *(const v8bf*)(&Al[ro + 16 + 8 * half]);
        a_l[i] = __builtin_shufflevector(c0, c1, 0,1,2,3,4,5,6,7,8,9,10,11,12,13,14,15);
      }
    }
    // B (16-bit 32x16 layout): lanes 0-15 -> K 0..7 & 16..23 (N=lane); lanes 16-31 -> K 8..15 & 24..31
#pragma unroll
    for (int j = 0; j < 4; ++j) {
      const int ro = (colBase + j * 16 + l16) * LSTR;
      {
        v8bf c0 = *(const v8bf*)(&Bh[ro + 8 * half]);
        v8bf c1 = *(const v8bf*)(&Bh[ro + 16 + 8 * half]);
        b_h[j] = __builtin_shufflevector(c0, c1, 0,1,2,3,4,5,6,7,8,9,10,11,12,13,14,15);
      }
      {
        v8bf c0 = *(const v8bf*)(&Bl[ro + 8 * half]);
        v8bf c1 = *(const v8bf*)(&Bl[ro + 16 + 8 * half]);
        b_l[j] = __builtin_shufflevector(c0, c1, 0,1,2,3,4,5,6,7,8,9,10,11,12,13,14,15);
      }
    }

    // ---- 3-product split-bf16 accumulation ----
#pragma unroll
    for (int i = 0; i < 2; ++i)
#pragma unroll
      for (int j = 0; j < 4; ++j) {
        acc[i][j] = __builtin_amdgcn_wmma_f32_16x16x32_bf16(
            false, a_h[i], false, b_l[j], (short)0, acc[i][j], false, false);
        acc[i][j] = __builtin_amdgcn_wmma_f32_16x16x32_bf16(
            false, a_l[i], false, b_h[j], (short)0, acc[i][j], false, false);
        acc[i][j] = __builtin_amdgcn_wmma_f32_16x16x32_bf16(
            false, a_h[i], false, b_h[j], (short)0, acc[i][j], false, false);
      }
    __syncthreads();
  }

  // epilogue: C/D layout -> lanes 0-15: (M=v, N=lane); lanes 16-31: (M=8+v, N=lane-16)
#pragma unroll
  for (int i = 0; i < 2; ++i)
#pragma unroll
    for (int j = 0; j < 4; ++j) {
      const size_t col = gCol0 + colBase + j * 16 + l16;
#pragma unroll
      for (int vv = 0; vv < 8; ++vv) {
        const size_t row = gRow0 + rowBase + i * 16 + half * 8 + vv;
        K[row * (size_t)n + col] = expf(EPS_INV * (acc[i][j][vv] - 1.0f));
      }
    }
}

// ---------------- u = 1 / (K @ v)  (one wave per row, float4 streaming) ----------------
__global__ void __launch_bounds__(256)
kv_recip(const float* __restrict__ K, const float* __restrict__ v,
         float* __restrict__ u, int n) {
  const int row  = blockIdx.x * 8 + (threadIdx.x >> 5);
  const int lane = threadIdx.x & 31;
  const float4* Kr = (const float4*)(K + (size_t)row * n);
  const float4* V4 = (const float4*)v;
  float acc = 0.0f;
  const int q = n >> 2;
  for (int it = lane; it < q; it += 32) {
    const float4 k4 = Kr[it];
    const float4 w4 = V4[it];
    acc += k4.x * w4.x + k4.y * w4.y + k4.z * w4.z + k4.w * w4.w;
  }
  acc = waveReduce(acc);
  if (lane == 0) u[row] = 1.0f / acc;
}

// ---------------- v = 1 / (K^T @ u): stage 1, partial over 128-row chunks ----------------
__global__ void __launch_bounds__(256)
ktu_partial(const float* __restrict__ K, const float* __restrict__ u,
            float* __restrict__ part, int n) {
  const int col = blockIdx.x * 256 + threadIdx.x;
  const int r0  = blockIdx.y * 128;
  float acc = 0.0f;
#pragma unroll 4
  for (int r = 0; r < 128; ++r)
    acc += K[(size_t)(r0 + r) * n + col] * u[r0 + r];
  part[(size_t)blockIdx.y * n + col] = acc;
}

// ---------------- v = 1 / (K^T @ u): stage 2, combine + reciprocal ----------------
__global__ void __launch_bounds__(256)
ktu_final(const float* __restrict__ part, float* __restrict__ v, int n, int chunks) {
  const int col = blockIdx.x * 256 + threadIdx.x;
  float acc = 0.0f;
  for (int p = 0; p < chunks; ++p) acc += part[(size_t)p * n + col];
  v[col] = 1.0f / acc;
}

// ---------------- loss = sum u_i * K_ij * v_j * C_ij,  C = -eps*ln(K) ----------------
__global__ void __launch_bounds__(256)
loss_partial(const float* __restrict__ K, const float* __restrict__ u,
             const float* __restrict__ v, float* __restrict__ part) {
  const size_t base = (size_t)blockIdx.x * 16384 + threadIdx.x;
  float acc = 0.0f;
#pragma unroll 4
  for (int k = 0; k < 64; ++k) {
    const size_t idx = base + (size_t)k * 256;
    const int i = (int)(idx >> 13);     // n == 8192
    const int j = (int)(idx & 8191);
    const float kk = K[idx];
    const float C = -EPS * logf(kk);
    acc += u[i] * kk * v[j] * C;
  }
  acc = blockReduce256(acc);
  if (threadIdx.x == 0) part[blockIdx.x] = acc;
}

__global__ void __launch_bounds__(256)
loss_final(const float* __restrict__ part, float* __restrict__ out, int nblocks) {
  float acc = 0.0f;
  for (int p = threadIdx.x; p < nblocks; p += 256) acc += part[p];
  acc = blockReduce256(acc);
  if (threadIdx.x == 0) out[0] = acc;
}

// ---------------- h = (alpha+beta)*xa + gamma*xt ----------------
__global__ void __launch_bounds__(256)
fuse_kernel(const float* __restrict__ xa, const float* __restrict__ xt,
            const float* __restrict__ alpha, const float* __restrict__ beta,
            const float* __restrict__ gamma, float* __restrict__ out) {
  const float ab = alpha[0] + beta[0];
  const float g  = gamma[0];
  const size_t idx = (size_t)blockIdx.x * 256 + threadIdx.x;  // float4 index
  const float4 a = ((const float4*)xa)[idx];
  const float4 t = ((const float4*)xt)[idx];
  float4 r;
  r.x = ab * a.x + g * t.x;
  r.y = ab * a.y + g * t.y;
  r.z = ab * a.z + g * t.z;
  r.w = ab * a.w + g * t.w;
  ((float4*)out)[idx] = r;
}

extern "C" void kernel_launch(void* const* d_in, const int* in_sizes, int n_in,
                              void* d_out, int out_size, void* d_ws, size_t ws_size,
                              hipStream_t stream) {
  const float* xa    = (const float*)d_in[0];
  const float* xt    = (const float*)d_in[1];
  const float* alpha = (const float*)d_in[2];
  const float* beta  = (const float*)d_in[3];
  const float* gamma = (const float*)d_in[4];
  const int N = in_sizes[0] / DIM;  // 8192
  float* out = (float*)d_out;

  // workspace layout
  char* ws = (char*)d_ws;
  size_t off = 0;
  float* K = (float*)(ws + off); off += (size_t)N * N * sizeof(float);  // 256 MB
  __bf16* xa_hi = (__bf16*)(ws + off); off += (size_t)N * DIM * sizeof(__bf16);  // 8 MB
  __bf16* xa_lo = (__bf16*)(ws + off); off += (size_t)N * DIM * sizeof(__bf16);
  __bf16* xt_hi = (__bf16*)(ws + off); off += (size_t)N * DIM * sizeof(__bf16);
  __bf16* xt_lo = (__bf16*)(ws + off); off += (size_t)N * DIM * sizeof(__bf16);
  float* u    = (float*)(ws + off); off += (size_t)N * sizeof(float);
  float* v    = (float*)(ws + off); off += (size_t)N * sizeof(float);
  float* part = (float*)(ws + off); off += (size_t)(N / 128) * N * sizeof(float);  // 2 MB
  float* lpart = (float*)(ws + off);                                               // 16 KB

  l2norm_split<<<2 * N, 256, 0, stream>>>(xa, xt, xa_hi, xa_lo, xt_hi, xt_lo, N);
  init_uv<<<(N + 255) / 256, 256, 0, stream>>>(u, v, N);

  gemm_exp_wmma<<<dim3(N / 128, N / 128), 256, 0, stream>>>(xa_hi, xa_lo, xt_hi, xt_lo, K, N);

  for (int it = 0; it < 5; ++it) {
    kv_recip<<<N / 8, 256, 0, stream>>>(K, v, u, N);
    ktu_partial<<<dim3(N / 256, N / 128), 256, 0, stream>>>(K, u, part, N);
    ktu_final<<<N / 256, 256, 0, stream>>>(part, v, N, N / 128);
  }

  const int lossBlocks = (int)(((size_t)N * N) / 16384);  // 4096
  loss_partial<<<lossBlocks, 256, 0, stream>>>(K, u, v, lpart);
  loss_final<<<1, 256, 0, stream>>>(lpart, out + (out_size - 1), lossBlocks);

  fuse_kernel<<<(N * DIM / 4) / 256, 256, 0, stream>>>(xa, xt, alpha, beta, gamma, out);
}
